// CDAMLayer_83013127897148
// MI455X (gfx1250) — compile-verified, matrix operand order checked by reading, not run on hardware
//
#include <hip/hip_runtime.h>
#include <math.h>

// ---------------- types ----------------
typedef unsigned short u16;
typedef __attribute__((ext_vector_type(16))) __bf16 v16bf;
typedef __attribute__((ext_vector_type(8)))  float  v8f;

union Frag { v16bf v; uint4 u[2]; };

#define TOK 65536           // total tokens (1024 windows * 64)
#define DIMC 256
#define HIDC 512

// ---------------- helpers ----------------
__device__ __forceinline__ u16 f2bf(float f) {
    unsigned u = __float_as_uint(f);
    unsigned r = (u + 0x7FFFu + ((u >> 16) & 1u)) >> 16;
    return (u16)r;
}
__device__ __forceinline__ unsigned pk2(float a, float b) {
    return (unsigned)f2bf(a) | ((unsigned)f2bf(b) << 16);
}
__device__ __forceinline__ uint4 pack8(const float* p) {
    float4 a = *(const float4*)p;
    float4 b = *(const float4*)(p + 4);
    uint4 r;
    r.x = pk2(a.x, a.y); r.y = pk2(a.z, a.w);
    r.z = pk2(b.x, b.y); r.w = pk2(b.z, b.w);
    return r;
}

// ---------------- f32 -> bf16 weight convert ----------------
__global__ __launch_bounds__(256) void cvt_bf16_kernel(const float* __restrict__ x,
                                                       u16* __restrict__ y, int n) {
    int i = blockIdx.x * 256 + threadIdx.x;
    if (i < n) y[i] = f2bf(x[i]);
}

// ---------------- relative position bias table: [8][64][64] ----------------
__global__ __launch_bounds__(256) void bias_kernel(const float* __restrict__ rpb,
                                                   float* __restrict__ tab) {
    int h = blockIdx.x;
    for (int idx = threadIdx.x; idx < 64 * 64; idx += 256) {
        int p1 = idx >> 6, p2 = idx & 63;
        int i1 = p1 >> 3, j1 = p1 & 7, i2 = p2 >> 3, j2 = p2 & 7;
        int rel = (i1 - i2 + 7) * 15 + (j1 - j2 + 7);
        tab[h * 4096 + idx] = rpb[rel * 8 + h];
    }
}

// ---------------- window partition / unpartition ----------------
// [B=4, C=256, H=128, W=128] <-> token-major [65536, 256]
__global__ __launch_bounds__(256) void part_kernel(const float* __restrict__ X,
                                                   float* __restrict__ T) {
    int win = blockIdx.x;                // 0..1023 : (b, wh, ww)
    int b = win >> 8, wh = (win >> 4) & 15, ww = win & 15;
    int c = threadIdx.x;
    const float* src = X + ((size_t)b * 256 + c) * 16384;
    float* dst = T + (size_t)win * 64 * DIMC + c;
    #pragma unroll 8
    for (int tok = 0; tok < 64; ++tok) {
        int i = tok >> 3, j = tok & 7;
        dst[(size_t)tok * DIMC] = src[(wh * 8 + i) * 128 + ww * 8 + j];
    }
}
__global__ __launch_bounds__(256) void unpart_kernel(const float* __restrict__ T,
                                                     float* __restrict__ X) {
    int win = blockIdx.x;
    int b = win >> 8, wh = (win >> 4) & 15, ww = win & 15;
    int c = threadIdx.x;
    float* dst = X + ((size_t)b * 256 + c) * 16384;
    const float* src = T + (size_t)win * 64 * DIMC + c;
    #pragma unroll 8
    for (int tok = 0; tok < 64; ++tok) {
        int i = tok >> 3, j = tok & 7;
        dst[(wh * 8 + i) * 128 + ww * 8 + j] = src[(size_t)tok * DIMC];
    }
}

// ---------------- LayerNorm over C=256: f32 in -> bf16 out ----------------
__global__ __launch_bounds__(256) void ln_kernel(const float* __restrict__ X,
                                                 const float* __restrict__ gg,
                                                 const float* __restrict__ bb,
                                                 u16* __restrict__ Y) {
    int wave = threadIdx.x >> 5, lane = threadIdx.x & 31;
    size_t tok = (size_t)blockIdx.x * 8 + wave;
    const float4* x4 = (const float4*)(X + tok * DIMC);
    float4 a = x4[lane];
    float4 c = x4[lane + 32];
    float s = a.x + a.y + a.z + a.w + c.x + c.y + c.z + c.w;
    #pragma unroll
    for (int o = 16; o; o >>= 1) s += __shfl_xor(s, o, 32);
    float mean = s * (1.0f / 256.0f);
    float dx0 = a.x - mean, dx1 = a.y - mean, dx2 = a.z - mean, dx3 = a.w - mean;
    float dy0 = c.x - mean, dy1 = c.y - mean, dy2 = c.z - mean, dy3 = c.w - mean;
    float v = dx0*dx0 + dx1*dx1 + dx2*dx2 + dx3*dx3 + dy0*dy0 + dy1*dy1 + dy2*dy2 + dy3*dy3;
    #pragma unroll
    for (int o = 16; o; o >>= 1) v += __shfl_xor(v, o, 32);
    float inv = rsqrtf(v * (1.0f / 256.0f) + 1e-5f);
    const float4* g4 = (const float4*)gg;
    const float4* b4 = (const float4*)bb;
    float4 ga = g4[lane], gc = g4[lane + 32], ba = b4[lane], bc = b4[lane + 32];
    uint2 r0, r1;
    r0.x = pk2(dx0 * inv * ga.x + ba.x, dx1 * inv * ga.y + ba.y);
    r0.y = pk2(dx2 * inv * ga.z + ba.z, dx3 * inv * ga.w + ba.w);
    r1.x = pk2(dy0 * inv * gc.x + bc.x, dy1 * inv * gc.y + bc.y);
    r1.y = pk2(dy2 * inv * gc.z + bc.z, dy3 * inv * gc.w + bc.w);
    uint2* y2 = (uint2*)(Y + tok * DIMC);
    y2[lane] = r0;
    y2[lane + 32] = r1;
}

// ---------------- generic bf16 WMMA GEMM: C = A[M,K] @ B[N,K]^T + bias ----------------
// EPI: 0 = store bf16, 1 = store bf16 * scale, 2 = gelu -> bf16, 3 = f32 residual add
template <int EPI>
__global__ __launch_bounds__(256) void gemm_bf16_kernel(const u16* __restrict__ A,
                                                        const u16* __restrict__ B,
                                                        const float* __restrict__ bias,
                                                        void* __restrict__ Cout,
                                                        float* __restrict__ Res,
                                                        int M, int N, int K, float scale) {
    int wave = threadIdx.x >> 5, lane = threadIdx.x & 31;
    int strips_n = N >> 4;
    int strip = blockIdx.x * 8 + wave;
    int m0 = (strip / strips_n) * 64;
    int n0 = (strip % strips_n) * 16;
    if (m0 >= M) return;               // wave-uniform
    int r = lane & 15, hi = lane >> 4;

    v8f acc[4] = {};
    for (int k = 0; k < K; k += 32) {
        Frag bf;
        const u16* bp = B + (size_t)(n0 + r) * K + k + hi * 8;
        bf.u[0] = *(const uint4*)bp;
        bf.u[1] = *(const uint4*)(bp + 16);
        #pragma unroll
        for (int mt = 0; mt < 4; ++mt) {
            Frag af;
            const u16* ap = A + (size_t)(m0 + mt * 16 + r) * K + k + hi * 8;
            af.u[0] = *(const uint4*)ap;
            af.u[1] = *(const uint4*)(ap + 16);
            acc[mt] = __builtin_amdgcn_wmma_f32_16x16x32_bf16(
                false, af.v, false, bf.v, (short)0, acc[mt], false, false);
        }
    }
    int n = n0 + r;
    float bn = bias[n];
    #pragma unroll
    for (int mt = 0; mt < 4; ++mt) {
        #pragma unroll
        for (int g = 0; g < 8; ++g) {
            int m = m0 + mt * 16 + g + hi * 8;
            float val = acc[mt][g] + bn;
            if (EPI == 1) val *= scale;
            if (EPI == 2) val = 0.5f * val * (1.0f + erff(val * 0.70710678118f));
            if (EPI == 3) {
                float* o = (float*)Cout;
                o[(size_t)m * N + n] = Res[(size_t)m * N + n] + val;
            } else {
                ((u16*)Cout)[(size_t)m * N + n] = f2bf(val);
            }
        }
    }
}

// ---------------- windowed cross attention (one wave = one head) ----------------
#define NPAD 68    // score row stride (floats)
#define VPAD 72    // v^T row stride (u16)

__global__ __launch_bounds__(64) void attn_kernel(const u16* __restrict__ Q,
                                                  const u16* __restrict__ KV,
                                                  const float* __restrict__ bias,
                                                  u16* __restrict__ Out) {
    __shared__ float sc[2][64 * NPAD];
    __shared__ u16 vt[2][32 * VPAD];
    int wave = threadIdx.x >> 5, lane = threadIdx.x & 31;
    int win = blockIdx.x;
    int head = blockIdx.y * 2 + wave;
    int t0 = win * 64;
    int cbase = head * 32;
    int r = lane & 15, hi = lane >> 4;
    int k0 = hi * 8;

    // Q (A-frags) and K (B-frags)
    Frag qa[4], kb[4];
    #pragma unroll
    for (int mt = 0; mt < 4; ++mt) {
        const u16* p = Q + (size_t)(t0 + mt * 16 + r) * DIMC + cbase + k0;
        qa[mt].u[0] = *(const uint4*)p;
        qa[mt].u[1] = *(const uint4*)(p + 16);
    }
    #pragma unroll
    for (int nt = 0; nt < 4; ++nt) {
        const u16* p = KV + (size_t)(t0 + nt * 16 + r) * HIDC + cbase + k0;
        kb[nt].u[0] = *(const uint4*)p;
        kb[nt].u[1] = *(const uint4*)(p + 16);
    }

    // scores = QK^T + bias -> LDS
    const float* bh = bias + head * 4096;
    float* S = sc[wave];
    #pragma unroll
    for (int mt = 0; mt < 4; ++mt) {
        #pragma unroll
        for (int nt = 0; nt < 4; ++nt) {
            v8f acc = {};
            acc = __builtin_amdgcn_wmma_f32_16x16x32_bf16(
                false, qa[mt].v, false, kb[nt].v, (short)0, acc, false, false);
            int n = nt * 16 + r;
            #pragma unroll
            for (int g = 0; g < 8; ++g) {
                int m = mt * 16 + g + hi * 8;
                S[m * NPAD + n] = acc[g] + bh[m * 64 + n];
            }
        }
    }
    __syncthreads();

    // row softmax in LDS (2 rows per lane)
    #pragma unroll
    for (int mm = 0; mm < 2; ++mm) {
        float* row = S + (mm * 32 + lane) * NPAD;
        float mx = -1e30f;
        for (int n = 0; n < 64; ++n) mx = fmaxf(mx, row[n]);
        float sum = 0.0f;
        for (int n = 0; n < 64; ++n) { float e = __expf(row[n] - mx); row[n] = e; sum += e; }
        float is = 1.0f / sum;
        for (int n = 0; n < 64; ++n) row[n] *= is;
    }

    // stage V^T: [32 ch][64 tok] bf16 in LDS
    u16* VT = vt[wave];
    for (int t = 0; t < 64; ++t)
        VT[lane * VPAD + t] = KV[(size_t)(t0 + t) * HIDC + 256 + cbase + lane];
    __syncthreads();

    // out = P @ V  (M=64 tok, N=32 ch, K=64 tok)
    #pragma unroll
    for (int mt = 0; mt < 4; ++mt) {
        Frag pa[2];
        #pragma unroll
        for (int kt = 0; kt < 2; ++kt) {
            const float* prow = S + (mt * 16 + r) * NPAD + kt * 32 + k0;
            pa[kt].u[0] = pack8(prow);
            pa[kt].u[1] = pack8(prow + 16);
        }
        #pragma unroll
        for (int ntv = 0; ntv < 2; ++ntv) {
            v8f acc = {};
            #pragma unroll
            for (int kt = 0; kt < 2; ++kt) {
                Frag vb;
                const u16* vp = VT + (ntv * 16 + r) * VPAD + kt * 32 + k0;
                vb.u[0] = *(const uint4*)vp;
                vb.u[1] = *(const uint4*)(vp + 16);
                acc = __builtin_amdgcn_wmma_f32_16x16x32_bf16(
                    false, pa[kt].v, false, vb.v, (short)0, acc, false, false);
            }
            #pragma unroll
            for (int g = 0; g < 8; ++g) {
                int m = t0 + mt * 16 + g + hi * 8;
                int n = cbase + ntv * 16 + r;
                Out[(size_t)m * DIMC + n] = f2bf(acc[g]);
            }
        }
    }
}

// ---------------- host ----------------
extern "C" void kernel_launch(void* const* d_in, const int* in_sizes, int n_in,
                              void* d_out, int out_size, void* d_ws, size_t ws_size,
                              hipStream_t stream) {
    const float* spatial   = (const float*)d_in[0];
    const float* frequency = (const float*)d_in[1];
    const float* s_qw = (const float*)d_in[2],  * s_qb  = (const float*)d_in[3];
    const float* s_kvw = (const float*)d_in[4], * s_kvb = (const float*)d_in[5];
    const float* s_rpb = (const float*)d_in[6];
    const float* s_pw = (const float*)d_in[7],  * s_pb  = (const float*)d_in[8];
    const float* f_qw = (const float*)d_in[9],  * f_qb  = (const float*)d_in[10];
    const float* f_kvw = (const float*)d_in[11],* f_kvb = (const float*)d_in[12];
    const float* f_rpb = (const float*)d_in[13];
    const float* f_pw = (const float*)d_in[14], * f_pb  = (const float*)d_in[15];
    const float* ns1_g = (const float*)d_in[16], * ns1_b = (const float*)d_in[17];
    const float* nfs_g = (const float*)d_in[18], * nfs_b = (const float*)d_in[19];
    const float* ns2_g = (const float*)d_in[20], * ns2_b = (const float*)d_in[21];
    const float* nf1_g = (const float*)d_in[22], * nf1_b = (const float*)d_in[23];
    const float* nsf_g = (const float*)d_in[24], * nsf_b = (const float*)d_in[25];
    const float* nf2_g = (const float*)d_in[26], * nf2_b = (const float*)d_in[27];
    const float* ffns_w1 = (const float*)d_in[28], * ffns_b1 = (const float*)d_in[29];
    const float* ffns_w2 = (const float*)d_in[30], * ffns_b2 = (const float*)d_in[31];
    const float* ffnf_w1 = (const float*)d_in[32], * ffnf_b1 = (const float*)d_in[33];
    const float* ffnf_w2 = (const float*)d_in[34], * ffnf_b2 = (const float*)d_in[35];

    float* outS = (float*)d_out;
    float* outF = outS + (size_t)4 * 256 * 128 * 128;

    const size_t MB = 1ull << 20;
    char* ws = (char*)d_ws;
    float* Swin = (float*)(ws + 0);          // 64 MB
    float* Fwin = (float*)(ws + 64 * MB);    // 64 MB
    u16* LNA = (u16*)(ws + 128 * MB);        // 32 MB
    u16* LNB = (u16*)(ws + 160 * MB);        // 32 MB
    u16* QB  = (u16*)(ws + 192 * MB);        // 32 MB
    u16* KVB = (u16*)(ws + 224 * MB);        // 64 MB  (also reused as MLP hidden)
    u16* AOB = (u16*)(ws + 288 * MB);        // 32 MB
    float* biasS = (float*)(ws + 320 * MB);  // 128 KB
    float* biasF = biasS + 8 * 4096;
    u16* wb = (u16*)(ws + 321 * MB);

    u16* s_qw_b = wb;    wb += 256 * 256;
    u16* s_kvw_b = wb;   wb += 512 * 256;
    u16* s_pw_b = wb;    wb += 256 * 256;
    u16* ffns_w1_b = wb; wb += 512 * 256;
    u16* ffns_w2_b = wb; wb += 256 * 512;
    u16* f_qw_b = wb;    wb += 256 * 256;
    u16* f_kvw_b = wb;   wb += 512 * 256;
    u16* f_pw_b = wb;    wb += 256 * 256;
    u16* ffnf_w1_b = wb; wb += 512 * 256;
    u16* ffnf_w2_b = wb; wb += 256 * 512;

    auto cvt = [&](const float* src, u16* dst, int n) {
        cvt_bf16_kernel<<<(n + 255) / 256, 256, 0, stream>>>(src, dst, n);
    };
    cvt(s_qw, s_qw_b, 256 * 256);      cvt(s_kvw, s_kvw_b, 512 * 256);
    cvt(s_pw, s_pw_b, 256 * 256);      cvt(ffns_w1, ffns_w1_b, 512 * 256);
    cvt(ffns_w2, ffns_w2_b, 256 * 512);
    cvt(f_qw, f_qw_b, 256 * 256);      cvt(f_kvw, f_kvw_b, 512 * 256);
    cvt(f_pw, f_pw_b, 256 * 256);      cvt(ffnf_w1, ffnf_w1_b, 512 * 256);
    cvt(ffnf_w2, ffnf_w2_b, 256 * 512);

    bias_kernel<<<8, 256, 0, stream>>>(s_rpb, biasS);
    bias_kernel<<<8, 256, 0, stream>>>(f_rpb, biasF);

    part_kernel<<<1024, 256, 0, stream>>>(spatial, Swin);
    part_kernel<<<1024, 256, 0, stream>>>(frequency, Fwin);

    auto gemm = [&](int epi, const u16* A, const u16* B, const float* bias, void* C,
                    float* res, int M, int N, int K, float scale) {
        int blocks = (M / 64) * (N / 16) / 8;
        switch (epi) {
            case 0: gemm_bf16_kernel<0><<<blocks, 256, 0, stream>>>(A, B, bias, C, res, M, N, K, scale); break;
            case 1: gemm_bf16_kernel<1><<<blocks, 256, 0, stream>>>(A, B, bias, C, res, M, N, K, scale); break;
            case 2: gemm_bf16_kernel<2><<<blocks, 256, 0, stream>>>(A, B, bias, C, res, M, N, K, scale); break;
            default: gemm_bf16_kernel<3><<<blocks, 256, 0, stream>>>(A, B, bias, C, res, M, N, K, scale); break;
        }
    };

    const float qscale = 0.17677669529663687f;   // 32^-0.5

    auto run_dir = [&](float* Xq, float* Xkv,
                       const float* g1, const float* b1, const float* gkv, const float* bkv,
                       u16* qwb, const float* qb, u16* kvwb, const float* kvb,
                       float* biasTab, u16* pwb, const float* pb,
                       const float* g2, const float* b2,
                       u16* w1b, const float* fb1, u16* w2b, const float* fb2) {
        ln_kernel<<<TOK / 8, 256, 0, stream>>>(Xq, g1, b1, LNA);
        ln_kernel<<<TOK / 8, 256, 0, stream>>>(Xkv, gkv, bkv, LNB);
        gemm(1, LNA, qwb, qb, QB, nullptr, TOK, 256, 256, qscale);
        gemm(0, LNB, kvwb, kvb, KVB, nullptr, TOK, 512, 256, 1.0f);
        attn_kernel<<<dim3(1024, 4), 64, 0, stream>>>(QB, KVB, biasTab, AOB);
        gemm(3, AOB, pwb, pb, Xq, Xq, TOK, 256, 256, 1.0f);       // residual add
        ln_kernel<<<TOK / 8, 256, 0, stream>>>(Xq, g2, b2, LNA);
        gemm(2, LNA, w1b, fb1, KVB, nullptr, TOK, 512, 256, 1.0f); // gelu, hidden reuses KVB
        gemm(3, KVB, w2b, fb2, Xq, Xq, TOK, 256, 512, 1.0f);      // residual add
    };

    // direction 1: spatial queries attend to frequency KV
    run_dir(Swin, Fwin, ns1_g, ns1_b, nfs_g, nfs_b,
            s_qw_b, s_qb, s_kvw_b, s_kvb, biasS, s_pw_b, s_pb,
            ns2_g, ns2_b, ffns_w1_b, ffns_b1, ffns_w2_b, ffns_b2);
    unpart_kernel<<<1024, 256, 0, stream>>>(Swin, outS);

    // direction 2: frequency queries attend to updated spatial KV
    run_dir(Fwin, Swin, nf1_g, nf1_b, nsf_g, nsf_b,
            f_qw_b, f_qb, f_kvw_b, f_kvb, biasF, f_pw_b, f_pb,
            nf2_g, nf2_b, ffnf_w1_b, ffnf_b1, ffnf_w2_b, ffnf_b2);
    unpart_kernel<<<1024, 256, 0, stream>>>(Fwin, outF);
}
